// BlipAttention_75007308857568
// MI455X (gfx1250) — matrix operand order repaired and checked by reading, not compiled, not run on hardware
//
#include <hip/hip_runtime.h>
#include <hip/hip_bf16.h>

#define EMBED 768
#define SEQ 1024
#define BATCH 32
#define M_TOTAL (BATCH * SEQ)   // 32768
#define QKV_N (3 * EMBED)       // 2304

typedef __attribute__((ext_vector_type(16))) __bf16 v16bf;
typedef __attribute__((ext_vector_type(8)))  float  v8f;
typedef unsigned int u32;
typedef __attribute__((ext_vector_type(4))) u32 u32x4;
typedef __attribute__((ext_vector_type(8))) u32 u32x8;

union FragBF16 { v16bf v; uint4 q[2]; };

// Load a 16-element bf16 fragment from two 16-byte-aligned 8-element chunks.
static __device__ inline v16bf frag_load2(const __bf16* p0, const __bf16* p1) {
  FragBF16 f;
  f.q[0] = *(const uint4*)p0;
  f.q[1] = *(const uint4*)p1;
  return f.v;
}

// Build an A fragment from fp32 memory (convert to bf16 in flight).
static __device__ inline v16bf frag_cvt_f32(const float* p0, const float* p1) {
  v16bf r;
#pragma unroll
  for (int i = 0; i < 8; ++i) r[i] = (__bf16)p0[i];
#pragma unroll
  for (int i = 0; i < 8; ++i) r[8 + i] = (__bf16)p1[i];
  return r;
}

static __device__ inline v8f wmma_bf16(v16bf a, v16bf b, v8f c) {
  // D = A(16x32 bf16) x B(32x16 bf16) + C(16x16 f32)
  return __builtin_amdgcn_wmma_f32_16x16x32_bf16(
      /*neg_a=*/false, a, /*neg_b=*/false, b,
      /*c_mod=*/(short)0, c, /*reuse_a=*/false, /*reuse_b=*/false);
}

// ---------------- Kernel 0: weight convert + transpose (W[K][N] -> WT[N][K] bf16)
__global__ __launch_bounds__(256) void wcvt_t(const float* __restrict__ W,
                                              __bf16* __restrict__ WT,
                                              int K, int N) {
  int idx = blockIdx.x * 256 + threadIdx.x;
  if (idx >= K * N) return;
  int k = idx / N, n = idx - k * N;
  WT[(size_t)n * K + k] = (__bf16)W[idx];
}

// ---------------- Kernel 1: QKV GEMM (+bias), writes Q,K row-major bf16, V transposed
__global__ __launch_bounds__(256) void qkv_gemm(const float* __restrict__ x,
                                                const __bf16* __restrict__ WT,
                                                const float* __restrict__ bias,
                                                __bf16* __restrict__ Qb,
                                                __bf16* __restrict__ Kb,
                                                __bf16* __restrict__ Vt) {
  const int lane = threadIdx.x & 31;
  const int wave = threadIdx.x >> 5;
  const int m0 = blockIdx.y * 128 + wave * 16;
  const int n0 = blockIdx.x * 64;
  const int abase = (lane & 16) ? 8 : 0;   // A: upper lane half holds K 8..15 / 24..31
  const int bofs  = (lane & 16) ? 8 : 0;   // B: upper lane half holds K 8..15 / 24..31
  const float* xrow = x + (size_t)(m0 + (lane & 15)) * EMBED;

  v8f acc[4] = {};
  for (int c0 = 0; c0 < EMBED; c0 += 32) {
    v16bf a = frag_cvt_f32(xrow + c0 + abase, xrow + c0 + abase + 16);
#pragma unroll
    for (int t = 0; t < 4; ++t) {
      const __bf16* bp = WT + (size_t)(n0 + t * 16 + (lane & 15)) * EMBED + c0 + bofs;
      acc[t] = wmma_bf16(a, frag_load2(bp, bp + 16), acc[t]);
    }
  }
  const int rowhi = (lane >> 4) << 3;
#pragma unroll
  for (int t = 0; t < 4; ++t) {
    int ncol = n0 + t * 16 + (lane & 15);
    float bv = bias[ncol];
    int sec = ncol / EMBED;           // 0=Q, 1=K, 2=V (uniform per tile)
    int nloc = ncol - sec * EMBED;
#pragma unroll
    for (int r = 0; r < 8; ++r) {
      int m = m0 + r + rowhi;
      __bf16 h = (__bf16)(acc[t][r] + bv);
      if (sec == 0)      Qb[(size_t)m * EMBED + nloc] = h;
      else if (sec == 1) Kb[(size_t)m * EMBED + nloc] = h;
      else {
        int bb = m >> 10, mq = m & 1023;
        Vt[((size_t)bb * EMBED + nloc) * SEQ + mq] = h;   // Vt[b][c][n]
      }
    }
  }
}

// ---------------- Kernel 2: fused attention. One block = (batch, 16 query rows).
// Q tile (16x768 bf16, 24KB) staged into LDS by the Tensor Data Mover.
// Phase 1: each wave computes S for 8 key tiles (full 16x1024 scores in regs).
// Softmax via shfl_xor + LDS cross-wave reduction; P -> LDS bf16.
// Phase 2: O = P @ V, each wave owns 6 C-tiles; store transposed OT[b][c][n].
__global__ __launch_bounds__(256) void attn_fused(const __bf16* __restrict__ Qb,
                                                  const __bf16* __restrict__ Kb,
                                                  const __bf16* __restrict__ Vt,
                                                  __bf16* __restrict__ OT) {
  __shared__ __bf16 Qs[16][EMBED];    // 24 KB, filled by TDM
  __shared__ __bf16 P[16][SEQ];       // 32 KB
  __shared__ float red[8][16];
  __shared__ float rowstat[16];

  const int b = blockIdx.x;
  const int q0 = blockIdx.y * 16;
  const int lane = threadIdx.x & 31;
  const int wave = threadIdx.x >> 5;
  const int abase = (lane & 16) ? 8 : 0;
  const int bofs  = (lane & 16) ? 8 : 0;
  const int rowhi = (lane >> 4) << 3;
  const float scale = 0.03608439182435161f;  // 1/sqrt(768)

  // ---- TDM: stage Q[b, q0:q0+16, 0:768] (row-major bf16) into Qs.
  if (wave == 0) {
    u32 ldsa = (u32)(uintptr_t)(&Qs[0][0]);
    unsigned long long ga =
        (unsigned long long)(uintptr_t)(Qb + ((size_t)b * SEQ + q0) * EMBED);
    u32x4 g0;
    g0[0] = 1u;                              // count=1 (valid user descriptor)
    g0[1] = ldsa;                            // lds_addr
    g0[2] = (u32)ga;                         // global_addr[31:0]
    g0[3] = (u32)(ga >> 32) | (2u << 30);    // global_addr[56:32] | type=2 (image)
    u32x8 g1;
    g1[0] = (1u << 16);                      // data_size=1 -> 2 bytes/elem
    g1[1] = ((u32)EMBED) << 16;              // tensor_dim0[15:0] = 768
    g1[2] = 16u << 16;                       // tensor_dim1[15:0] = 16
    g1[3] = ((u32)EMBED) << 16;              // tile_dim0 = 768
    g1[4] = 16u;                             // tile_dim1 = 16 (tile_dim2 = 0)
    g1[5] = (u32)EMBED;                      // tensor_dim0_stride[31:0] = 768
    g1[6] = 0u;
    g1[7] = 0u;
    asm volatile("tensor_load_to_lds %0, %1" :: "s"(g0), "s"(g1) : "memory");
    __builtin_amdgcn_s_wait_tensorcnt(0);
  }
  __syncthreads();

  const __bf16* qrow = &Qs[lane & 15][0];

  v8f s[8] = {};
  for (int c0 = 0; c0 < EMBED; c0 += 32) {
    v16bf a = frag_load2(qrow + c0 + abase, qrow + c0 + abase + 16);
#pragma unroll
    for (int j = 0; j < 8; ++j) {
      int kt = wave + j * 8;  // key tile 0..63
      const __bf16* kp = Kb + ((size_t)b * SEQ + kt * 16 + (lane & 15)) * EMBED + c0 + bofs;
      s[j] = wmma_bf16(a, frag_load2(kp, kp + 16), s[j]);
    }
  }
#pragma unroll
  for (int j = 0; j < 8; ++j)
#pragma unroll
    for (int r = 0; r < 8; ++r) s[j][r] *= scale;

  // ---- row max (per VGPR r: row = rowhi + r)
  float rowm[8];
#pragma unroll
  for (int r = 0; r < 8; ++r) {
    float m = s[0][r];
#pragma unroll
    for (int j = 1; j < 8; ++j) m = fmaxf(m, s[j][r]);
    m = fmaxf(m, __shfl_xor(m, 1, 32));
    m = fmaxf(m, __shfl_xor(m, 2, 32));
    m = fmaxf(m, __shfl_xor(m, 4, 32));
    m = fmaxf(m, __shfl_xor(m, 8, 32));
    rowm[r] = m;
  }
  if ((lane & 15) == 0) {
#pragma unroll
    for (int r = 0; r < 8; ++r) red[wave][rowhi + r] = rowm[r];
  }
  __syncthreads();
  if (threadIdx.x < 16) {
    float g = red[0][threadIdx.x];
#pragma unroll
    for (int w = 1; w < 8; ++w) g = fmaxf(g, red[w][threadIdx.x]);
    rowstat[threadIdx.x] = g;
  }
  __syncthreads();
#pragma unroll
  for (int r = 0; r < 8; ++r) rowm[r] = rowstat[rowhi + r];

  // ---- exp + row sum
  float rinv[8];
#pragma unroll
  for (int r = 0; r < 8; ++r) {
    float sum = 0.f;
#pragma unroll
    for (int j = 0; j < 8; ++j) {
      float e = __expf(s[j][r] - rowm[r]);
      s[j][r] = e;
      sum += e;
    }
    sum += __shfl_xor(sum, 1, 32);
    sum += __shfl_xor(sum, 2, 32);
    sum += __shfl_xor(sum, 4, 32);
    sum += __shfl_xor(sum, 8, 32);
    rinv[r] = sum;
  }
  __syncthreads();  // all rowstat(max) reads done before red/rowstat reuse
  if ((lane & 15) == 0) {
#pragma unroll
    for (int r = 0; r < 8; ++r) red[wave][rowhi + r] = rinv[r];
  }
  __syncthreads();
  if (threadIdx.x < 16) {
    float g = 0.f;
#pragma unroll
    for (int w = 0; w < 8; ++w) g += red[w][threadIdx.x];
    rowstat[threadIdx.x] = 1.0f / g;
  }
  __syncthreads();
#pragma unroll
  for (int r = 0; r < 8; ++r) rinv[r] = rowstat[rowhi + r];

  // ---- write normalized P to LDS (row-major [16][1024] bf16)
#pragma unroll
  for (int j = 0; j < 8; ++j) {
    int col = (wave + j * 8) * 16 + (lane & 15);
#pragma unroll
    for (int r = 0; r < 8; ++r)
      P[rowhi + r][col] = (__bf16)(s[j][r] * rinv[r]);
  }
  __syncthreads();

  // ---- Phase 2: O = P(16x1024) @ V(1024x768), wave owns C-tiles {wave, wave+8, ...}
  v8f o[6] = {};
  for (int kk = 0; kk < SEQ; kk += 32) {
    const __bf16* pr = &P[lane & 15][kk + abase];
    v16bf a = frag_load2(pr, pr + 16);
#pragma unroll
    for (int t = 0; t < 6; ++t) {
      int cc = (wave + t * 8) * 16 + (lane & 15);
      const __bf16* vp = Vt + ((size_t)b * EMBED + cc) * SEQ + kk + bofs;
      o[t] = wmma_bf16(a, frag_load2(vp, vp + 16), o[t]);
    }
  }
#pragma unroll
  for (int t = 0; t < 6; ++t) {
    int cc = (wave + t * 8) * 16 + (lane & 15);
#pragma unroll
    for (int r = 0; r < 8; ++r) {
      int row = q0 + r + rowhi;
      OT[((size_t)b * EMBED + cc) * SEQ + row] = (__bf16)o[t][r];  // OT[b][c][n]
    }
  }
}

// ---------------- Kernel 3: proj GEMM. The reference's swapaxes+reshape is a flat
// reinterpretation of OT[b][c][n]: permuted A row n', col k = OT_b flat[n'*768 + k].
__global__ __launch_bounds__(256) void proj_gemm(const __bf16* __restrict__ OT,
                                                 const __bf16* __restrict__ WT,
                                                 const float* __restrict__ bias,
                                                 float* __restrict__ out) {
  const int lane = threadIdx.x & 31;
  const int wave = threadIdx.x >> 5;
  const int m0 = blockIdx.y * 128 + wave * 16;
  const int n0 = blockIdx.x * 64;
  const int m = m0 + (lane & 15);
  const int abase = (lane & 16) ? 8 : 0;
  const int bofs  = (lane & 16) ? 8 : 0;
  const __bf16* obase = OT + ((size_t)(m >> 10) * EMBED * SEQ) + (size_t)(m & 1023) * EMBED;

  v8f acc[4] = {};
  for (int c0 = 0; c0 < EMBED; c0 += 32) {
    const __bf16* p0 = obase + c0 + abase;
    v16bf a = frag_load2(p0, p0 + 16);
#pragma unroll
    for (int t = 0; t < 4; ++t) {
      const __bf16* bp = WT + (size_t)(n0 + t * 16 + (lane & 15)) * EMBED + c0 + bofs;
      acc[t] = wmma_bf16(a, frag_load2(bp, bp + 16), acc[t]);
    }
  }
  const int rowhi = (lane >> 4) << 3;
#pragma unroll
  for (int t = 0; t < 4; ++t) {
    int ncol = n0 + t * 16 + (lane & 15);
    float bv = bias[ncol];
#pragma unroll
    for (int r = 0; r < 8; ++r) {
      int mm = m0 + r + rowhi;
      out[(size_t)mm * EMBED + ncol] = acc[t][r] + bv;
    }
  }
}

extern "C" void kernel_launch(void* const* d_in, const int* in_sizes, int n_in,
                              void* d_out, int out_size, void* d_ws, size_t ws_size,
                              hipStream_t stream) {
  const float* x      = (const float*)d_in[0];
  const float* qkv_w  = (const float*)d_in[1];
  const float* qkv_b  = (const float*)d_in[2];
  const float* proj_w = (const float*)d_in[3];
  const float* proj_b = (const float*)d_in[4];
  float* out = (float*)d_out;

  char* ws = (char*)d_ws;
  const size_t SZ = (size_t)M_TOTAL * EMBED * 2;  // one bf16 activation buffer = 48 MB
  __bf16* Qb     = (__bf16*)(ws);
  __bf16* Kb     = (__bf16*)(ws + SZ);
  __bf16* Vt     = (__bf16*)(ws + 2 * SZ);
  __bf16* OTb    = (__bf16*)(ws + 3 * SZ);
  __bf16* WqkvT  = (__bf16*)(ws + 4 * SZ);
  __bf16* WprojT = (__bf16*)(ws + 4 * SZ + (size_t)QKV_N * EMBED * 2);
  // total ws usage ~196.5 MB

  wcvt_t<<<dim3((EMBED * QKV_N + 255) / 256), 256, 0, stream>>>(qkv_w, WqkvT, EMBED, QKV_N);
  wcvt_t<<<dim3((EMBED * EMBED + 255) / 256), 256, 0, stream>>>(proj_w, WprojT, EMBED, EMBED);
  qkv_gemm<<<dim3(QKV_N / 64, M_TOTAL / 128), 256, 0, stream>>>(x, WqkvT, qkv_b, Qb, Kb, Vt);
  attn_fused<<<dim3(BATCH, SEQ / 16), 256, 0, stream>>>(Qb, Kb, Vt, OTb);
  proj_gemm<<<dim3(EMBED / 64, M_TOTAL / 128), 256, 0, stream>>>(OTb, WprojT, proj_b, out);
}
